// Local_Global_Attention_de1_16956530885138
// MI455X (gfx1250) — matrix operand order repaired and checked by reading, not compiled, not run on hardware
//
#include <hip/hip_runtime.h>

typedef __attribute__((ext_vector_type(16))) _Float16 v16h;
typedef __attribute__((ext_vector_type(8)))  float    v8f;
typedef __attribute__((ext_vector_type(4)))  float    f4v;

// ---------------- problem constants ----------------
#define NB 16
#define NN 307
#define TT 48
#define DDIM 256
#define HD 32

// LDS row strides (in halves / floats), padded for bank spread on b128 frag loads
#define RS_IN 264   // 48 x 256 f16 input staging
#define RS_P  136   // 48 x 128 f16 projection outputs
#define RS_VT 72    // 128 x 64 f16 transposed V (time padded 48->64 with zeros)
#define RS_S  49    // 48 x 48 f32 scores (per wave)
#define RS_PB 72    // 48 x 64 f16 probabilities (per wave)

// LDS byte offsets
#define OFF_Q   0        // 48*264*2 = 25344
#define OFF_K   25344
#define OFF_V   50688
#define OFF_TQ  76032    // 48*136*2 = 13056
#define OFF_TK  89088
#define OFF_LQ  102144
#define OFF_LK  115200
#define OFF_TVT 128256   // 128*72*2 = 18432
#define OFF_LVT 146688
#define OFF_S   165120   // 8 * 48*49*4 = 75264
#define OFF_P   240384   // 8 * 48*72*2 = 55296
#define SMEM_BYTES 295680
#define OFF_CAT 0        // reuse sQ region after projections: 48 x 256 f16 concat buffer

// f16 weight layout inside d_ws (offsets in halves)
#define WTQ 0
#define WTK 32768
#define WTV 65536
#define WW1 98304
#define WQC 131072      // 3 x (128x256)
#define WKC 229376      // 3 x (128x256)
#define WW2 327680      // 256x256
#define WTOTAL 393216

union FragU  { v16h h; uint4 u[2]; };
union Pack4  { _Float16 h[4]; uint2 u; };
union Pack8  { _Float16 h[8]; uint4 u; };

__device__ __forceinline__ v8f wmma_f16(v16h a, v16h b, v8f c) {
  return __builtin_amdgcn_wmma_f32_16x16x32_f16(false, a, false, b, (short)0, c, false, false);
}

// A fragment: 16x32 f16 tile, row-major, row = M, per ISA A layout.
__device__ __forceinline__ v16h load_fragA(const _Float16* base, int rs) {
  const int l  = threadIdx.x & 15;
  const int hi = (threadIdx.x >> 4) & 1;
  const _Float16* p = base + l * rs + hi * 8;
  FragU f;
  f.u[0] = *(const uint4*)(p);
  f.u[1] = *(const uint4*)(p + 16);
  return f.h;
}

// A fragment from a per-lane row pointer, value-masked (branch-free zero padding).
__device__ __forceinline__ v16h load_fragA_mask(const _Float16* rowbase, int col0, unsigned m) {
  const int hi = (threadIdx.x >> 4) & 1;
  const _Float16* p = rowbase + col0 + hi * 8;
  FragU f;
  f.u[0] = *(const uint4*)(p);
  f.u[1] = *(const uint4*)(p + 16);
  f.u[0].x &= m; f.u[0].y &= m; f.u[0].z &= m; f.u[0].w &= m;
  f.u[1].x &= m; f.u[1].y &= m; f.u[1].z &= m; f.u[1].w &= m;
  return f.h;
}

// B fragment: loads B^T stored row-major (row = N col of B, cols = K), per ISA B striping.
__device__ __forceinline__ v16h load_fragB(const _Float16* base, int rs) {
  const int l  = threadIdx.x & 15;
  const int hi = (threadIdx.x >> 4) & 1;
  const _Float16* p = base + l * rs + hi * 8;
  FragU f;
  f.u[0] = *(const uint4*)(p);
  f.u[1] = *(const uint4*)(p + 16);
  return f.h;
}

// Per-lane clamped row index + 0/~0 mask for segment-aware temporal shift.
__device__ __forceinline__ void shift_row_mask(int t, int shift, bool seg,
                                               int& tpc, unsigned& m) {
  const int tp = t + shift;
  int s0 = 0, s1 = TT;
  if (seg) {
    s0 = (t < 12) ? 0  : ((t < 24) ? 12 : 24);
    s1 = (t < 12) ? 12 : ((t < 24) ? 24 : 48);
  }
  const bool ok = (tp >= s0) && (tp < s1);
  int c = tp < 0 ? 0 : tp;
  tpc = c > 47 ? 47 : c;
  m = ok ? 0xffffffffu : 0u;
}

// ---------------- weight prep: f32 -> f16, de-stride conv kernels ----------------
__global__ __launch_bounds__(256) void prep_weights(
    const float* __restrict__ tq, const float* __restrict__ tk, const float* __restrict__ tv,
    const float* __restrict__ w1, const float* __restrict__ qc, const float* __restrict__ kc,
    const float* __restrict__ w2, _Float16* __restrict__ ws) {
  const int i = blockIdx.x * 256 + threadIdx.x;
  const int M = 32768;
  if (i >= WTOTAL) return;
  float v;
  if      (i <  M)     v = tq[i];
  else if (i < 2 * M)  v = tk[i - M];
  else if (i < 3 * M)  v = tv[i - 2 * M];
  else if (i < 4 * M)  v = w1[i - 3 * M];
  else if (i < 7 * M)  { int r = i - 4 * M; int k = r / M; int od = r - k * M; v = qc[od * 3 + k]; }
  else if (i < 10 * M) { int r = i - 7 * M; int k = r / M; int od = r - k * M; v = kc[od * 3 + k]; }
  else                 v = w2[i - 10 * M];
  ws[i] = (_Float16)v;
}

// ---------------- fused local-global attention, one (b,n) per workgroup ----------------
__global__ __launch_bounds__(256) void lga_main(
    const float* __restrict__ q, const float* __restrict__ kx, const float* __restrict__ vv,
    const _Float16* __restrict__ ws,
    const float* __restrict__ tq_b, const float* __restrict__ tk_b, const float* __restrict__ tv_b,
    const float* __restrict__ qc_b, const float* __restrict__ kc_b,
    const float* __restrict__ b1,  const float* __restrict__ b2,
    const int* __restrict__ qms_p, const int* __restrict__ kms_p,
    float* __restrict__ out) {
  extern __shared__ char smem[];
  _Float16* sQ   = (_Float16*)(smem + OFF_Q);
  _Float16* sK   = (_Float16*)(smem + OFF_K);
  _Float16* sV   = (_Float16*)(smem + OFF_V);
  _Float16* sTQ  = (_Float16*)(smem + OFF_TQ);
  _Float16* sTK  = (_Float16*)(smem + OFF_TK);
  _Float16* sLQ  = (_Float16*)(smem + OFF_LQ);
  _Float16* sLK  = (_Float16*)(smem + OFF_LK);
  _Float16* sTVt = (_Float16*)(smem + OFF_TVT);
  _Float16* sLVt = (_Float16*)(smem + OFF_LVT);
  float*    Sall = (float*)   (smem + OFF_S);
  _Float16* Pall = (_Float16*)(smem + OFF_P);
  _Float16* sCAT = (_Float16*)(smem + OFF_CAT);

  const int tid  = threadIdx.x;
  // Provably wave-uniform wave index -> scalar loop control / scalar job selection.
  const int wave = __builtin_amdgcn_readfirstlane(tid) >> 5;
  const int l    = tid & 15;
  const int hi   = (tid >> 4) & 1;
  const long long bn = blockIdx.x;

  const float* qsrc = q  + bn * (TT * DDIM);
  const float* ksrc = kx + bn * (TT * DDIM);
  const float* vsrc = vv + bn * (TT * DDIM);

  // ---- stage 1: stage Q/K/V into LDS as f16 (non-temporal reads), zero V^T tails ----
  for (int idx = tid; idx < 3072; idx += 256) {
    const int t = idx >> 6;
    const int c = (idx & 63) << 2;
    f4v a;
    Pack4 p;
    a = __builtin_nontemporal_load((const f4v*)(qsrc + t * DDIM + c));
    p.h[0] = (_Float16)a.x; p.h[1] = (_Float16)a.y; p.h[2] = (_Float16)a.z; p.h[3] = (_Float16)a.w;
    *(uint2*)(sQ + t * RS_IN + c) = p.u;
    a = __builtin_nontemporal_load((const f4v*)(ksrc + t * DDIM + c));
    p.h[0] = (_Float16)a.x; p.h[1] = (_Float16)a.y; p.h[2] = (_Float16)a.z; p.h[3] = (_Float16)a.w;
    *(uint2*)(sK + t * RS_IN + c) = p.u;
    a = __builtin_nontemporal_load((const f4v*)(vsrc + t * DDIM + c));
    p.h[0] = (_Float16)a.x; p.h[1] = (_Float16)a.y; p.h[2] = (_Float16)a.z; p.h[3] = (_Float16)a.w;
    *(uint2*)(sV + t * RS_IN + c) = p.u;
  }
  for (int idx = tid; idx < 4096; idx += 256) {      // zero time cols [48,64) of V^T buffers
    _Float16* buf = (idx >> 11) ? sLVt : sTVt;
    const int r = (idx >> 4) & 127;
    const int c = idx & 15;
    buf[r * RS_VT + 48 + c] = (_Float16)0.f;
  }
  __syncthreads();

  // ---- stage 2: six 48x256x128 matmuls (projections + segment-aware shifted convs) ----
  const int qmsv = qms_p[0];
  const int kmsv = kms_p[0];
  const bool lq_seg  = (qmsv != 0) && (kmsv != 0);
  const bool lk_seg  = (qmsv != 0) || (kmsv != 0);
  const _Float16* lk_src = ((qmsv == 0) && (kmsv == 0)) ? sQ : sK;

  for (int cj = wave; cj < 48; cj += 8) {
    const int job = cj >> 3;          // scalar
    const int tj  = cj & 7;           // scalar
    const _Float16* src;  const _Float16* wbase; const float* bias;
    _Float16* dst = nullptr; _Float16* dstT = nullptr;
    int nk = 1, shoff = 0; bool seg = false;
    switch (job) {
      case 0:  src = sQ;     wbase = ws + WTQ; bias = tq_b; dst  = sTQ;  break;
      case 1:  src = sK;     wbase = ws + WTK; bias = tk_b; dst  = sTK;  break;
      case 2:  src = sQ;     wbase = ws + WQC; bias = qc_b; dst  = sLQ;  nk = 3; shoff = -2; seg = lq_seg; break;
      case 3:  src = lk_src; wbase = ws + WKC; bias = kc_b; dst  = sLK;  nk = 3; shoff = -1; seg = lk_seg; break;
      case 4:  src = sV;     wbase = ws + WTV; bias = tv_b; dstT = sTVt; break;
      default: src = sV;     wbase = ws + WW1; bias = b1;   dstT = sLVt; break;
    }
    v8f acc0 = {}, acc1 = {}, acc2 = {};
    for (int k = 0; k < nk; ++k) {
      const _Float16* wk = wbase + k * 32768;
      if (nk == 3) {                  // scalar branch; branch-free masked A loads inside
        const int shift = shoff + k;
        int tp0, tp1, tp2; unsigned m0, m1, m2;
        shift_row_mask(l,      shift, seg, tp0, m0);
        shift_row_mask(16 + l, shift, seg, tp1, m1);
        shift_row_mask(32 + l, shift, seg, tp2, m2);
        const _Float16* r0 = src + tp0 * RS_IN;
        const _Float16* r1 = src + tp1 * RS_IN;
        const _Float16* r2 = src + tp2 * RS_IN;
        for (int kk = 0; kk < 8; ++kk) {
          const v16h bf = load_fragB(wk + (tj * 16) * 256 + kk * 32, 256);
          const v16h a0 = load_fragA_mask(r0, kk * 32, m0);
          const v16h a1 = load_fragA_mask(r1, kk * 32, m1);
          const v16h a2 = load_fragA_mask(r2, kk * 32, m2);
          acc0 = wmma_f16(a0, bf, acc0);
          acc1 = wmma_f16(a1, bf, acc1);
          acc2 = wmma_f16(a2, bf, acc2);
        }
      } else {
        for (int kk = 0; kk < 8; ++kk) {
          const v16h bf = load_fragB(wk + (tj * 16) * 256 + kk * 32, 256);
          const v16h a0 = load_fragA(src + 0  * RS_IN + kk * 32, RS_IN);
          const v16h a1 = load_fragA(src + 16 * RS_IN + kk * 32, RS_IN);
          const v16h a2 = load_fragA(src + 32 * RS_IN + kk * 32, RS_IN);
          acc0 = wmma_f16(a0, bf, acc0);
          acc1 = wmma_f16(a1, bf, acc1);
          acc2 = wmma_f16(a2, bf, acc2);
        }
      }
    }
    const float bv = bias[tj * 16 + l];
    if (dstT) {    // transposed (time-minor) store: one packed b128 per row-tile per lane
      Pack8 p0, p1, p2;
      for (int r = 0; r < 8; ++r) {
        p0.h[r] = (_Float16)(acc0[r] + bv);
        p1.h[r] = (_Float16)(acc1[r] + bv);
        p2.h[r] = (_Float16)(acc2[r] + bv);
      }
      _Float16* dp = dstT + (tj * 16 + l) * RS_VT + hi * 8;
      *(uint4*)(dp + 0)  = p0.u;
      *(uint4*)(dp + 16) = p1.u;
      *(uint4*)(dp + 32) = p2.u;
    } else {
      for (int r = 0; r < 8; ++r) {
        dst[(0  + r + hi * 8) * RS_P + tj * 16 + l] = (_Float16)(acc0[r] + bv);
        dst[(16 + r + hi * 8) * RS_P + tj * 16 + l] = (_Float16)(acc1[r] + bv);
        dst[(32 + r + hi * 8) * RS_P + tj * 16 + l] = (_Float16)(acc2[r] + bv);
      }
    }
  }
  __syncthreads();

  // ---- stage 3: attention, one head per wave (waves 0-3 global, 4-7 local) ----
  {
    const int  h   = wave & 3;        // scalar
    const bool isg = (wave < 4);      // scalar
    const _Float16* Qm = isg ? sTQ  : sLQ;
    const _Float16* Km = isg ? sTK  : sLK;
    const _Float16* Vt = isg ? sTVt : sLVt;
    float*    Sb = Sall + wave * (48 * RS_S);
    _Float16* Pb = Pall + wave * (48 * RS_PB);
    const int ch = h * HD;
    const float SCALE = 0.17677669529663687f;   // 1/sqrt(32)

    const v16h qf0 = load_fragA(Qm + 0  * RS_P + ch, RS_P);
    const v16h qf1 = load_fragA(Qm + 16 * RS_P + ch, RS_P);
    const v16h qf2 = load_fragA(Qm + 32 * RS_P + ch, RS_P);
#pragma unroll
    for (int jk = 0; jk < 3; ++jk) {
      const v16h kf = load_fragB(Km + (jk * 16) * RS_P + ch, RS_P);
      v8f s0 = {}, s1 = {}, s2 = {};
      s0 = wmma_f16(qf0, kf, s0);
      s1 = wmma_f16(qf1, kf, s1);
      s2 = wmma_f16(qf2, kf, s2);
      for (int r = 0; r < 8; ++r) {
        Sb[(0  + r + hi * 8) * RS_S + jk * 16 + l] = s0[r] * SCALE;
        Sb[(16 + r + hi * 8) * RS_S + jk * 16 + l] = s1[r] * SCALE;
        Sb[(32 + r + hi * 8) * RS_S + jk * 16 + l] = s2[r] * SCALE;
      }
    }
    // softmax over rows (per-wave scratch; same-wave LDS ops are in-order)
    const int lane = tid & 31;
    for (int rr = lane; rr < 48; rr += 32) {
      float mx = -3.0e38f;
      for (int c = 0; c < 48; ++c) mx = fmaxf(mx, Sb[rr * RS_S + c]);
      float sum = 0.f;
      for (int c = 0; c < 48; ++c) {
        const float e = __expf(Sb[rr * RS_S + c] - mx);
        Sb[rr * RS_S + c] = e;
        sum += e;
      }
      const float inv = 1.f / sum;
      for (int c = 0; c < 48; ++c) Pb[rr * RS_PB + c] = (_Float16)(Sb[rr * RS_S + c] * inv);
      for (int c = 48; c < 64; ++c) Pb[rr * RS_PB + c] = (_Float16)0.f;
    }
    // P @ V : 48x64 (padded) x 64x32, V^T fragments are contiguous in sTVt/sLVt
    const int cbase = (isg ? 0 : 128) + ch;
#pragma unroll
    for (int jv = 0; jv < 2; ++jv) {
      v8f o0 = {}, o1 = {}, o2 = {};
#pragma unroll
      for (int kk = 0; kk < 2; ++kk) {
        const v16h bf = load_fragB(Vt + (ch + jv * 16) * RS_VT + kk * 32, RS_VT);
        const v16h p0 = load_fragA(Pb + 0  * RS_PB + kk * 32, RS_PB);
        const v16h p1 = load_fragA(Pb + 16 * RS_PB + kk * 32, RS_PB);
        const v16h p2 = load_fragA(Pb + 32 * RS_PB + kk * 32, RS_PB);
        o0 = wmma_f16(p0, bf, o0);
        o1 = wmma_f16(p1, bf, o1);
        o2 = wmma_f16(p2, bf, o2);
      }
      for (int r = 0; r < 8; ++r) {
        sCAT[(0  + r + hi * 8) * RS_IN + cbase + jv * 16 + l] = (_Float16)o0[r];
        sCAT[(16 + r + hi * 8) * RS_IN + cbase + jv * 16 + l] = (_Float16)o1[r];
        sCAT[(32 + r + hi * 8) * RS_IN + cbase + jv * 16 + l] = (_Float16)o2[r];
      }
    }
  }
  __syncthreads();

  // ---- stage 4: output projection 48x256 @ w2^T (256x256) + b2 -> global (NT stores) ----
  {
    const _Float16* wsW2 = ws + WW2;
    float* obase = out + bn * (TT * DDIM);
    for (int tj = wave; tj < 16; tj += 8) {
      v8f acc0 = {}, acc1 = {}, acc2 = {};
      for (int kk = 0; kk < 8; ++kk) {
        const v16h bf = load_fragB(wsW2 + (tj * 16) * 256 + kk * 32, 256);
        const v16h a0 = load_fragA(sCAT + 0  * RS_IN + kk * 32, RS_IN);
        const v16h a1 = load_fragA(sCAT + 16 * RS_IN + kk * 32, RS_IN);
        const v16h a2 = load_fragA(sCAT + 32 * RS_IN + kk * 32, RS_IN);
        acc0 = wmma_f16(a0, bf, acc0);
        acc1 = wmma_f16(a1, bf, acc1);
        acc2 = wmma_f16(a2, bf, acc2);
      }
      const float bv = b2[tj * 16 + l];
      for (int r = 0; r < 8; ++r) {
        __builtin_nontemporal_store(acc0[r] + bv, &obase[(0  + r + hi * 8) * DDIM + tj * 16 + l]);
        __builtin_nontemporal_store(acc1[r] + bv, &obase[(16 + r + hi * 8) * DDIM + tj * 16 + l]);
        __builtin_nontemporal_store(acc2[r] + bv, &obase[(32 + r + hi * 8) * DDIM + tj * 16 + l]);
      }
    }
  }
}

extern "C" void kernel_launch(void* const* d_in, const int* in_sizes, int n_in,
                              void* d_out, int out_size, void* d_ws, size_t ws_size,
                              hipStream_t stream) {
  const float* q    = (const float*)d_in[0];
  const float* kx   = (const float*)d_in[1];
  const float* v    = (const float*)d_in[2];
  const float* tq_w = (const float*)d_in[3];
  const float* tq_b = (const float*)d_in[4];
  const float* tk_w = (const float*)d_in[5];
  const float* tk_b = (const float*)d_in[6];
  const float* tv_w = (const float*)d_in[7];
  const float* tv_b = (const float*)d_in[8];
  const float* qc_w = (const float*)d_in[9];
  const float* qc_b = (const float*)d_in[10];
  const float* kc_w = (const float*)d_in[11];
  const float* kc_b = (const float*)d_in[12];
  const float* w1   = (const float*)d_in[13];
  const float* b1   = (const float*)d_in[14];
  const float* w2   = (const float*)d_in[15];
  const float* b2   = (const float*)d_in[16];
  const int*   qms  = (const int*)d_in[17];
  const int*   kms  = (const int*)d_in[18];
  _Float16* ws16 = (_Float16*)d_ws;
  float* out = (float*)d_out;
  (void)in_sizes; (void)n_in; (void)out_size; (void)ws_size;

  prep_weights<<<(WTOTAL + 255) / 256, 256, 0, stream>>>(tq_w, tk_w, tv_w, w1, qc_w, kc_w, w2, ws16);
  lga_main<<<NB * NN, 256, SMEM_BYTES, stream>>>(q, kx, v, ws16,
                                                 tq_b, tk_b, tv_b, qc_b, kc_b, b1, b2,
                                                 qms, kms, out);
}
